// S4Layer_2662879724372
// MI455X (gfx1250) — matrix operand order repaired and checked by reading, not compiled, not run on hardware
//
#include <hip/hip_runtime.h>
#include <hip/hip_bf16.h>

#define D_MODEL 1024
#define D_STATE 64
#define BATCH   16
#define LENGTH  4096
#define LC      64
#define NCHUNK  (LENGTH / LC)

typedef __attribute__((ext_vector_type(16))) __bf16 v16bf;
typedef __attribute__((ext_vector_type(8)))  float  v8f;

__device__ __forceinline__ __bf16 f2bf(float x) { return (__bf16)x; }

// A-fragment (16x32 bf16): per lane, 16 consecutive K values [kb..kb+15].
// p points at row + k0 + kb. Non-temporal: u is a 268MB single-use stream.
__device__ __forceinline__ v16bf load_a_frag_nt(const float* __restrict__ p) {
    v16bf a;
    #pragma unroll
    for (int e = 0; e < 8; e++) {
        a[e]     = f2bf(__builtin_nontemporal_load(p + e));
        a[e + 8] = f2bf(__builtin_nontemporal_load(p + 8 + e));
    }
    return a;
}

// A-fragment from an L2-resident source (xs, 17MB): regular temporal loads.
__device__ __forceinline__ v16bf load_a_frag(const float* __restrict__ p) {
    v16bf a;
    #pragma unroll
    for (int e = 0; e < 8; e++) { a[e] = f2bf(p[e]); a[e + 8] = f2bf(p[8 + e]); }
    return a;
}

// B-fragment (32x16 bf16): per lane, 16 consecutive K values (cached, reused).
__device__ __forceinline__ v16bf load_b_frag(const __bf16* __restrict__ p) {
    v16bf b;
    #pragma unroll
    for (int e = 0; e < 16; e++) b[e] = p[e];
    return b;
}

// ---------------------------------------------------------------------------
// K1: dt_mean reduction, bf16 conversions of dt*B and C, powers of a_bar^T.
// One block, 1024 threads.
// ---------------------------------------------------------------------------
__global__ void prep_kernel(const float* __restrict__ A,
                            const float* __restrict__ Bmat,
                            const float* __restrict__ Cmat,
                            const float* __restrict__ log_dt,
                            float* __restrict__ dt_mean_out,
                            float* __restrict__ Ppow,      // [LC][64][64], Ppow[k] = (a_bar^T)^(k+1)
                            __bf16* __restrict__ Bs,       // [64][1024] bf16 = dt*B
                            __bf16* __restrict__ Cb)       // [1024][64] bf16 = C
{
    __shared__ float red[1024];
    __shared__ float P1[D_STATE * D_STATE];
    __shared__ float ping[D_STATE * D_STATE];
    __shared__ float pong[D_STATE * D_STATE];

    const int tid = threadIdx.x;

    red[tid] = expf(log_dt[tid]);
    __syncthreads();
    for (int s = 512; s > 0; s >>= 1) {
        if (tid < s) red[tid] += red[tid + s];
        __syncthreads();
    }
    const float dt = red[0] * (1.0f / 1024.0f);
    if (tid == 0) dt_mean_out[0] = dt;

    for (int i = tid; i < D_STATE * D_MODEL; i += 1024) Bs[i] = f2bf(dt * Bmat[i]);
    for (int i = tid; i < D_MODEL * D_STATE; i += 1024) Cb[i] = f2bf(Cmat[i]);

    // P1[m][n] = a_bar[n][m] = delta(m,n) + dt*A[n][m]   (a_bar^T)
    for (int i = tid; i < D_STATE * D_STATE; i += 1024) {
        const int m = i >> 6, n = i & 63;
        const float v = dt * A[n * D_STATE + m] + (m == n ? 1.0f : 0.0f);
        P1[i] = v; ping[i] = v; Ppow[i] = v;
    }
    __syncthreads();

    float* src = ping;
    float* dst = pong;
    for (int k = 1; k < LC; k++) {
        for (int i = tid; i < D_STATE * D_STATE; i += 1024) {
            const int m = i >> 6, n = i & 63;
            float acc = 0.0f;
            #pragma unroll 8
            for (int j = 0; j < D_STATE; j++)
                acc += src[m * D_STATE + j] * P1[j * D_STATE + n];
            dst[i] = acc;
            Ppow[(size_t)k * D_STATE * D_STATE + i] = acc;
        }
        __syncthreads();
        float* t = src; src = dst; dst = t;
    }
}

// ---------------------------------------------------------------------------
// K2: bu[l][b][n] = sum_d u[b][l][d] * (dt*B)[n][d]   (bf16 WMMA, f32 acc)
// One wave per (b, 16-row l-tile); 16x64 output, K=1024.
// K loop unrolled by 2 with true ping-pong register sets: loads land directly
// in the slot consumed one half-step later -> no inter-buffer copies.
// ---------------------------------------------------------------------------
__global__ void gemm_bu_kernel(const float* __restrict__ u,
                               const __bf16* __restrict__ Bs,
                               float* __restrict__ bu)
{
    const int wave = blockIdx.x * (blockDim.x >> 5) + (threadIdx.x >> 5);
    const int lane = threadIdx.x & 31;
    const int b  = wave >> 8;     // 0..15
    const int lt = wave & 255;    // 0..255
    const int l0 = lt * 16;

    const int mr  = lane & 15;
    const int hi  = (lane >> 4) & 1;
    const int kbA = hi ? 16 : 0;
    const int kbB = hi ? 16 : 0;
    const int nB  = lane & 15;

    const float*  abase = u + ((size_t)b * LENGTH + l0 + mr) * D_MODEL + kbA;
    const __bf16* bbase = Bs + (size_t)nB * D_MODEL + kbB;

    v8f acc[4] = {};

    v16bf aA = load_a_frag_nt(abase);
    v16bf bA[4];
    #pragma unroll
    for (int nt = 0; nt < 4; nt++)
        bA[nt] = load_b_frag(bbase + (size_t)nt * 16 * D_MODEL);

    for (int k0 = 0; k0 < D_MODEL - 64; k0 += 64) {
        v16bf aB = load_a_frag_nt(abase + k0 + 32);
        v16bf bB[4];
        #pragma unroll
        for (int nt = 0; nt < 4; nt++)
            bB[nt] = load_b_frag(bbase + (size_t)nt * 16 * D_MODEL + k0 + 32);
        #pragma unroll
        for (int nt = 0; nt < 4; nt++)
            acc[nt] = __builtin_amdgcn_wmma_f32_16x16x32_bf16(
                false, aA, false, bA[nt], (short)0, acc[nt], false, false);

        aA = load_a_frag_nt(abase + k0 + 64);
        #pragma unroll
        for (int nt = 0; nt < 4; nt++)
            bA[nt] = load_b_frag(bbase + (size_t)nt * 16 * D_MODEL + k0 + 64);
        #pragma unroll
        for (int nt = 0; nt < 4; nt++)
            acc[nt] = __builtin_amdgcn_wmma_f32_16x16x32_bf16(
                false, aB, false, bB[nt], (short)0, acc[nt], false, false);
    }
    // tail: k-steps at D_MODEL-64 (in aA/bA) and D_MODEL-32
    {
        v16bf aB = load_a_frag_nt(abase + D_MODEL - 32);
        v16bf bB[4];
        #pragma unroll
        for (int nt = 0; nt < 4; nt++)
            bB[nt] = load_b_frag(bbase + (size_t)nt * 16 * D_MODEL + D_MODEL - 32);
        #pragma unroll
        for (int nt = 0; nt < 4; nt++)
            acc[nt] = __builtin_amdgcn_wmma_f32_16x16x32_bf16(
                false, aA, false, bA[nt], (short)0, acc[nt], false, false);
        #pragma unroll
        for (int nt = 0; nt < 4; nt++)
            acc[nt] = __builtin_amdgcn_wmma_f32_16x16x32_bf16(
                false, aB, false, bB[nt], (short)0, acc[nt], false, false);
    }

    #pragma unroll
    for (int nt = 0; nt < 4; nt++) {
        const int n = nt * 16 + nB;
        #pragma unroll
        for (int v = 0; v < 8; v++) {
            const int ll = l0 + v + (hi ? 8 : 0);
            bu[((size_t)ll * BATCH + b) * D_STATE + n] = acc[nt][v];
        }
    }
}

// ---------------------------------------------------------------------------
// K3: per-chunk local scan from zero state (in-place over bu -> z).
// One block per chunk; thread (b,n); ping-pong state, 1 barrier per step.
// ---------------------------------------------------------------------------
__global__ void local_scan_kernel(const float* __restrict__ Ppow,
                                  float* __restrict__ bu,
                                  float* __restrict__ chunk_end)
{
    __shared__ float AT[D_STATE * D_STATE];
    __shared__ float xb[2][BATCH][D_STATE];
    const int tid = threadIdx.x;
    const int b = tid >> 6, n = tid & 63;
    const int c = blockIdx.x;

    for (int i = tid; i < D_STATE * D_STATE; i += 1024) AT[i] = Ppow[i]; // (a_bar^T)^1
    xb[0][b][n] = 0.0f;
    __syncthreads();

    int cur = 0;
    float acc = 0.0f;
    for (int t = 0; t < LC; t++) {
        const size_t idx = (((size_t)(c * LC + t)) * BATCH + b) * D_STATE + n;
        acc = bu[idx];
        #pragma unroll 8
        for (int m = 0; m < D_STATE; m++) acc += xb[cur][b][m] * AT[m * D_STATE + n];
        xb[cur ^ 1][b][n] = acc;       // other buffer: no hazard with readers of cur
        bu[idx] = acc;
        __syncthreads();
        cur ^= 1;
    }
    chunk_end[((size_t)c * BATCH + b) * D_STATE + n] = acc;
}

// ---------------------------------------------------------------------------
// K4: serial prefix over chunks: sstart[c] = sstart[c-1] @ M + zend[c-1],
// M = (a_bar^T)^LC. One block, ping-pong state.
// ---------------------------------------------------------------------------
__global__ void prefix_kernel(const float* __restrict__ Ppow,
                              const float* __restrict__ chunk_end,
                              float* __restrict__ sstart)
{
    __shared__ float M[D_STATE * D_STATE];
    __shared__ float sb[2][BATCH][D_STATE];
    const int tid = threadIdx.x;
    const int b = tid >> 6, n = tid & 63;

    for (int i = tid; i < D_STATE * D_STATE; i += 1024)
        M[i] = Ppow[(size_t)(LC - 1) * D_STATE * D_STATE + i];
    sb[0][b][n] = 0.0f;
    __syncthreads();

    int cur = 0;
    for (int c = 0; c < NCHUNK; c++) {
        const size_t base = ((size_t)c * BATCH + b) * D_STATE + n;
        sstart[base] = sb[cur][b][n];
        float acc = chunk_end[base];
        #pragma unroll 8
        for (int m = 0; m < D_STATE; m++) acc += sb[cur][b][m] * M[m * D_STATE + n];
        sb[cur ^ 1][b][n] = acc;
        __syncthreads();
        cur ^= 1;
    }
}

// ---------------------------------------------------------------------------
// K5: fix-up: xs_t = z_t + sstart[c] @ (a_bar^T)^(k+1). In-place over bu.
// One block per chunk; P[k] staged through LDS.
// ---------------------------------------------------------------------------
__global__ void fixup_kernel(const float* __restrict__ Ppow,
                             const float* __restrict__ sstart,
                             float* __restrict__ bu)
{
    __shared__ float s[BATCH][D_STATE];
    __shared__ float Pk[D_STATE * D_STATE];
    const int tid = threadIdx.x;
    const int b = tid >> 6, n = tid & 63;
    const int c = blockIdx.x;

    s[b][n] = sstart[((size_t)c * BATCH + b) * D_STATE + n];
    for (int k = 0; k < LC; k++) {
        __syncthreads();
        for (int i = tid; i < D_STATE * D_STATE; i += 1024)
            Pk[i] = Ppow[(size_t)k * D_STATE * D_STATE + i];
        __syncthreads();
        float acc = 0.0f;
        #pragma unroll 8
        for (int m = 0; m < D_STATE; m++) acc += s[b][m] * Pk[m * D_STATE + n];
        const size_t idx = (((size_t)(c * LC + k)) * BATCH + b) * D_STATE + n;
        bu[idx] += acc;
    }
}

// ---------------------------------------------------------------------------
// K6: y[b][l][d] = sum_n xs[l][b][n] * C[d][n] + u[b][l][d]*D[d]
// One wave per (l, 64-wide d-tile), K=64: all fragments loaded up front,
// then 8 back-to-back WMMAs. u reads / y writes are non-temporal streams.
// ---------------------------------------------------------------------------
__global__ void gemm_y_kernel(const float* __restrict__ xs,
                              const __bf16* __restrict__ Cb,
                              const float* __restrict__ u,
                              const float* __restrict__ Dvec,
                              float* __restrict__ y)
{
    const int wave = blockIdx.x * (blockDim.x >> 5) + (threadIdx.x >> 5);
    const int lane = threadIdx.x & 31;
    const int l  = wave >> 4;          // 0..4095
    const int d0 = (wave & 15) * 64;   // 0..960

    const int mr  = lane & 15;         // batch index (A-row)
    const int hi  = (lane >> 4) & 1;
    const int kbA = hi ? 16 : 0;
    const int kbB = hi ? 16 : 0;
    const int nB  = lane & 15;

    const float*  arow  = xs + ((size_t)l * BATCH + mr) * D_STATE + kbA;
    const __bf16* bbase = Cb + (size_t)(d0 + nB) * D_STATE + kbB;

    // K=64 -> two k-steps, all fragments loaded before any WMMA
    v16bf a0 = load_a_frag(arow);
    v16bf a1 = load_a_frag(arow + 32);
    v16bf b0[4], b1[4];
    #pragma unroll
    for (int nt = 0; nt < 4; nt++) {
        const __bf16* bp = bbase + (size_t)nt * 16 * D_STATE;
        b0[nt] = load_b_frag(bp);
        b1[nt] = load_b_frag(bp + 32);
    }

    v8f acc[4] = {};
    #pragma unroll
    for (int nt = 0; nt < 4; nt++)
        acc[nt] = __builtin_amdgcn_wmma_f32_16x16x32_bf16(
            false, a0, false, b0[nt], (short)0, acc[nt], false, false);
    #pragma unroll
    for (int nt = 0; nt < 4; nt++)
        acc[nt] = __builtin_amdgcn_wmma_f32_16x16x32_bf16(
            false, a1, false, b1[nt], (short)0, acc[nt], false, false);

    #pragma unroll
    for (int nt = 0; nt < 4; nt++) {
        const int d = d0 + nt * 16 + nB;
        const float dv = Dvec[d];
        #pragma unroll
        for (int v = 0; v < 8; v++) {
            const int b = v + (hi ? 8 : 0);
            const size_t idx = ((size_t)b * LENGTH + l) * D_MODEL + d;
            const float uv = __builtin_nontemporal_load(u + idx);
            __builtin_nontemporal_store(acc[nt][v] + uv * dv, y + idx);
        }
    }
}

// ---------------------------------------------------------------------------
extern "C" void kernel_launch(void* const* d_in, const int* in_sizes, int n_in,
                              void* d_out, int out_size, void* d_ws, size_t ws_size,
                              hipStream_t stream)
{
    (void)in_sizes; (void)n_in; (void)out_size; (void)ws_size;

    const float* u      = (const float*)d_in[0];
    const float* A      = (const float*)d_in[1];
    const float* B      = (const float*)d_in[2];
    const float* C      = (const float*)d_in[3];
    const float* Dvec   = (const float*)d_in[4];
    const float* log_dt = (const float*)d_in[5];
    float* y = (float*)d_out;

    char* ws = (char*)d_ws;
    size_t off = 0;
    auto alloc = [&](size_t bytes) -> void* {
        void* p = ws + off;
        off = (off + bytes + 255) & ~(size_t)255;
        return p;
    };

    float*  dt_mean   = (float*)alloc(sizeof(float));
    float*  Ppow      = (float*)alloc((size_t)LC * D_STATE * D_STATE * sizeof(float));
    __bf16* Bs        = (__bf16*)alloc((size_t)D_STATE * D_MODEL * sizeof(__bf16));
    __bf16* Cb        = (__bf16*)alloc((size_t)D_MODEL * D_STATE * sizeof(__bf16));
    float*  chunk_end = (float*)alloc((size_t)NCHUNK * BATCH * D_STATE * sizeof(float));
    float*  sstart    = (float*)alloc((size_t)NCHUNK * BATCH * D_STATE * sizeof(float));
    float*  bu        = (float*)alloc((size_t)LENGTH * BATCH * D_STATE * sizeof(float));

    prep_kernel<<<1, 1024, 0, stream>>>(A, B, C, log_dt, dt_mean, Ppow, Bs, Cb);
    gemm_bu_kernel<<<512, 256, 0, stream>>>(u, Bs, bu);            // 4096 waves
    local_scan_kernel<<<NCHUNK, 1024, 0, stream>>>(Ppow, bu, chunk_end);
    prefix_kernel<<<1, 1024, 0, stream>>>(Ppow, chunk_end, sstart);
    fixup_kernel<<<NCHUNK, 1024, 0, stream>>>(Ppow, sstart, bu);
    gemm_y_kernel<<<8192, 256, 0, stream>>>(bu, Cb, u, Dvec, y);   // 65536 waves
}
